// SDTv3Attention_37692632989895
// MI455X (gfx1250) — matrix operand order repaired and checked by reading, not compiled, not run on hardware
//
#include <hip/hip_runtime.h>

typedef __attribute__((ext_vector_type(4)))  _Float16 v4h;
typedef __attribute__((ext_vector_type(8)))  _Float16 v8h;
typedef __attribute__((ext_vector_type(16))) _Float16 v16h;
typedef __attribute__((ext_vector_type(8)))  float    v8f;

#define DEV static __device__ __forceinline__

// ---- gfx1250 async global->LDS copy (ASYNCcnt-tracked), guarded fallback ----
#if defined(__has_builtin)
#  if __has_builtin(__builtin_amdgcn_global_load_async_to_lds_b128)
#    define USE_ASYNC_LDS 1
#  endif
#endif
#ifndef USE_ASYNC_LDS
#  define USE_ASYNC_LDS 0
#endif

#if USE_ASYNC_LDS
typedef int v4i_b __attribute__((vector_size(16)));
typedef __attribute__((address_space(1))) v4i_b gv4i;  // global (AS1)
typedef __attribute__((address_space(3))) v4i_b lv4i;  // LDS (AS3)
#endif

DEV void async_copy16(const _Float16* g, _Float16* l) {
#if USE_ASYNC_LDS
  __builtin_amdgcn_global_load_async_to_lds_b128((gv4i*)g, (lv4i*)l, 0, 0);
#else
  *(v8h*)l = *(const v8h*)g;
#endif
}

DEV void async_wait_all() {
#if USE_ASYNC_LDS
#  if __has_builtin(__builtin_amdgcn_s_wait_asynccnt)
  __builtin_amdgcn_s_wait_asynccnt(0);
#  else
  asm volatile("s_wait_asynccnt 0x0" ::: "memory");
#  endif
#endif
}

DEV v8f v8f_zero() { v8f z = {0.f,0.f,0.f,0.f,0.f,0.f,0.f,0.f}; return z; }

DEV v16h frag_combine(v8h lo, v8h hi) {
  v16h f;
#pragma unroll
  for (int i = 0; i < 8; ++i) { f[i] = lo[i]; f[i + 8] = hi[i]; }
  return f;
}

// A fragment (16 rows x 32 K), source row-major, leading dim ld (halves).
// ISA 16-bit A layout: lanes 0-15 row=lane, K chunks {0..7, 16..23};
//                      lanes 16-31 row=lane-16, K chunks {8..15, 24..31}.
DEV v16h load_fragA(const _Float16* base, int ld) {
  const int lane = threadIdx.x & 31;
  const _Float16* p = base + (lane & 15) * ld + ((lane >> 4) << 3);
  return frag_combine(*(const v8h*)p, *(const v8h*)(p + 16));
}

// B fragment (32 K x 16 cols), source stored transposed ([col][K] row-major).
// Lanes 0-15: col=lane, K chunks {0..7, 16..23}; lanes 16-31: col=lane-16, K chunks {8..15, 24..31}.
DEV v16h load_fragB(const _Float16* base, int ld) {
  const int lane = threadIdx.x & 31;
  const _Float16* p = base + (lane & 15) * ld + ((lane >> 4) << 3);
  return frag_combine(*(const v8h*)p, *(const v8h*)(p + 16));
}

DEV v8f wmma16x16x32(v16h a, v16h b, v8f c) {
  return __builtin_amdgcn_wmma_f32_16x16x32_f16(false, a, false, b, (short)0, c,
                                                false, false);
}

// ---------------------------------------------------------------------------
// Elementwise prep kernels
// ---------------------------------------------------------------------------
__global__ void f32_to_f16_kernel(const float* __restrict__ in,
                                  _Float16* __restrict__ out, size_t n) {
  size_t i = ((size_t)blockIdx.x * 256 + threadIdx.x) * 4;
  if (i + 4 <= n) {
    float4 f = *(const float4*)&in[i];
    v4h h; h[0] = (_Float16)f.x; h[1] = (_Float16)f.y;
    h[2] = (_Float16)f.z;        h[3] = (_Float16)f.w;
    *(v4h*)&out[i] = h;
  } else {
    for (; i < n; ++i) out[i] = (_Float16)in[i];
  }
}

__global__ void bn_coef_kernel(const float* __restrict__ g, const float* __restrict__ b,
                               const float* __restrict__ m, const float* __restrict__ v,
                               float* __restrict__ sc, float* __restrict__ bo, int n) {
  int i = blockIdx.x * 256 + threadIdx.x;
  if (i < n) {
    float s = g[i] * rsqrtf(v[i] + 1e-5f);
    sc[i] = s;
    bo[i] = b[i] - m[i] * s;
  }
}

// ---------------------------------------------------------------------------
// LIF scan over T (tau=2, vth=1, hard reset). In-place: spikes overwrite input.
// Loads issued up-front for MLP; the scan itself is 4 serial FMAs.
// ---------------------------------------------------------------------------
__global__ void lif_inplace_f16_kernel(_Float16* __restrict__ buf, size_t S) {
  size_t i = (size_t)blockIdx.x * 256 + threadIdx.x;
  if (i >= S) return;
  float x[4];
#pragma unroll
  for (int t = 0; t < 4; ++t) x[t] = (float)buf[t * S + i];
  float v = 0.f;
#pragma unroll
  for (int t = 0; t < 4; ++t) {
    v += (x[t] - v) * 0.5f;
    float s = (v >= 1.0f) ? 1.0f : 0.0f;
    buf[t * S + i] = (_Float16)s;
    v *= (1.0f - s);
  }
}

__global__ void lif_inplace_f32_kernel(float* __restrict__ buf, size_t S) {
  size_t i = (size_t)blockIdx.x * 256 + threadIdx.x;
  if (i >= S) return;
  float x[4];
#pragma unroll
  for (int t = 0; t < 4; ++t) x[t] = buf[t * S + i];
  float v = 0.f;
#pragma unroll
  for (int t = 0; t < 4; ++t) {
    v += (x[t] - v) * 0.5f;
    float s = (v >= 1.0f) ? 1.0f : 0.0f;
    buf[t * S + i] = s;
    v *= (1.0f - s);
  }
}

// ---------------------------------------------------------------------------
// Tiled GEMM: C[m,n] = sum_k A[m,k] * W[n,k], fused BN epilogue.
// Block tile 128x128, BK=32, 256 threads = 8 waves; wave tile 64x32 (4x2 WMMA).
// Double-buffered LDS fed by async global->LDS copies (ASYNCcnt).
// ---------------------------------------------------------------------------
template <bool OUT_F16>
__global__ __launch_bounds__(256) void gemm_bn_kernel(
    const _Float16* __restrict__ A,   // [M, K] row-major
    const _Float16* __restrict__ W,   // [Nc, K] row-major (acts as B^T)
    const float* __restrict__ sc, const float* __restrict__ bo,
    void* __restrict__ Cout, int Nc, int K) {
  __shared__ __align__(16) _Float16 sA[2][128 * 40];
  __shared__ __align__(16) _Float16 sB[2][128 * 40];
  const int tid = threadIdx.x;
  const int bn0 = blockIdx.x * 128;
  const size_t bm0 = (size_t)blockIdx.y * 128;
  const int w = tid >> 5;
  const int wm = (w & 1) * 64;   // wave row offset in block tile
  const int wn = (w >> 1) * 32;  // wave col offset in block tile

  // Each thread stages one contiguous 32B span per operand tile:
  // row = tid>>1 (0..127), col = (tid&1)*16 halves.
  const int lrow = tid >> 1;
  const int lcol = (tid & 1) * 16;
  const _Float16* gA = &A[(bm0 + lrow) * (size_t)K + lcol];
  const _Float16* gW = &W[(size_t)(bn0 + lrow) * K + lcol];

  auto stage = [&](int buf, int k0) {
    _Float16* la = &sA[buf][lrow * 40 + lcol];
    _Float16* lw = &sB[buf][lrow * 40 + lcol];
    async_copy16(gA + k0, la);
    async_copy16(gA + k0 + 8, la + 8);
    async_copy16(gW + k0, lw);
    async_copy16(gW + k0 + 8, lw + 8);
  };

  v8f acc[4][2];
#pragma unroll
  for (int i = 0; i < 4; ++i)
#pragma unroll
    for (int j = 0; j < 2; ++j) acc[i][j] = v8f_zero();

  const int nsteps = K / 32;
  stage(0, 0);
  for (int s = 0; s < nsteps; ++s) {
    const int cur = s & 1;
    async_wait_all();          // tile `cur` has landed in LDS
    __syncthreads();
    if (s + 1 < nsteps) stage(cur ^ 1, (s + 1) * 32);  // prefetch next tile

    v16h af[4], bf[2];
#pragma unroll
    for (int i = 0; i < 4; ++i) af[i] = load_fragA(&sA[cur][(wm + i * 16) * 40], 40);
#pragma unroll
    for (int j = 0; j < 2; ++j) bf[j] = load_fragB(&sB[cur][(wn + j * 16) * 40], 40);
#pragma unroll
    for (int i = 0; i < 4; ++i)
#pragma unroll
      for (int j = 0; j < 2; ++j) acc[i][j] = wmma16x16x32(af[i], bf[j], acc[i][j]);
    __syncthreads();
  }

  const int lane = tid & 31;
  const int colL = lane & 15;
  const int rsel = (lane >> 4) * 8;
#pragma unroll
  for (int j = 0; j < 2; ++j) {
    int col = bn0 + wn + j * 16 + colL;
    float s = sc[col], o = bo[col];
#pragma unroll
    for (int i = 0; i < 4; ++i) {
      size_t rbase = bm0 + wm + i * 16 + rsel;
#pragma unroll
      for (int g = 0; g < 8; ++g) {
        float val = acc[i][j][g] * s + o;
        if (OUT_F16)
          ((_Float16*)Cout)[(rbase + g) * Nc + col] = (_Float16)val;
        else
          ((float*)Cout)[(rbase + g) * Nc + col] = val;
      }
    }
  }
}

// ---------------------------------------------------------------------------
// Linear attention per (t,b,h): kv = k^T v (64x64, reduce N=1024),
// attn = q @ kv * D^-0.5, fused attn-BN. Spikes are {0,1} -> f16 exact.
// ---------------------------------------------------------------------------
__global__ __launch_bounds__(256) void attn_kernel(
    const _Float16* __restrict__ qkv,  // [T*B, N, 3C] spikes
    const float* __restrict__ scA, const float* __restrict__ boA,
    _Float16* __restrict__ attn) {     // [T*B, N, C] pre-LIF
  constexpr int Nn = 1024, C = 512, Dh = 64, LD3 = 1536;
  __shared__ __align__(16) _Float16 sKT[64 * 40];  // k^T tile: [d][n-chunk]
  __shared__ __align__(16) _Float16 sVT[64 * 40];  // v^T tile: [e][n-chunk]
  __shared__ __align__(16) _Float16 sKV[64 * 72];  // kv^T: [e][d]

  const int tb = blockIdx.x >> 3;
  const int h = blockIdx.x & 7;
  const _Float16* qbase = qkv + (size_t)tb * Nn * LD3 + h * Dh;
  const _Float16* kbase = qbase + C;
  const _Float16* vbase = qbase + 2 * C;
  const int tid = threadIdx.x, w = tid >> 5, lane = tid & 31;
  const int colL = lane & 15, rsel = (lane >> 4) * 8;

  // ---- phase 1: kv = k^T v ; 16 16x16 tiles, 2 per wave ----
  int ti[2] = {w * 2, w * 2 + 1};
  v8f pacc[2];
  pacc[0] = v8f_zero(); pacc[1] = v8f_zero();

  const int nrow = tid >> 3;          // 0..31
  const int dc = (tid & 7) * 8;       // 0..56
  for (int n0 = 0; n0 < Nn; n0 += 32) {
    v8h kk = *(const v8h*)&kbase[(size_t)(n0 + nrow) * LD3 + dc];
    v8h vv = *(const v8h*)&vbase[(size_t)(n0 + nrow) * LD3 + dc];
#pragma unroll
    for (int ii = 0; ii < 8; ++ii) {  // transposed LDS store
      sKT[(dc + ii) * 40 + nrow] = kk[ii];
      sVT[(dc + ii) * 40 + nrow] = vv[ii];
    }
    __syncthreads();
#pragma unroll
    for (int u = 0; u < 2; ++u) {
      int i = ti[u] >> 2, j = ti[u] & 3;
      v16h a = load_fragA(&sKT[i * 16 * 40], 40);
      v16h b = load_fragB(&sVT[j * 16 * 40], 40);
      pacc[u] = wmma16x16x32(a, b, pacc[u]);
    }
    __syncthreads();
  }
  // write kv transposed into LDS: sKV[e][d]
#pragma unroll
  for (int u = 0; u < 2; ++u) {
    int i = ti[u] >> 2, j = ti[u] & 3;
#pragma unroll
    for (int g = 0; g < 8; ++g)
      sKV[(j * 16 + colL) * 72 + i * 16 + rsel + g] = (_Float16)pacc[u][g];
  }
  __syncthreads();

  // ---- phase 2: attn = q @ kv, scale + BN ----
  for (int mt = w; mt < 64; mt += 8) {
    v8f am[4];
#pragma unroll
    for (int j = 0; j < 4; ++j) am[j] = v8f_zero();
    const _Float16* qrow = qbase + (size_t)mt * 16 * LD3;
#pragma unroll
    for (int ks = 0; ks < 64; ks += 32) {
      v16h af = load_fragA(qrow + ks, LD3);
#pragma unroll
      for (int j = 0; j < 4; ++j) {
        v16h bf = load_fragB(&sKV[j * 16 * 72 + ks], 72);
        am[j] = wmma16x16x32(af, bf, am[j]);
      }
    }
#pragma unroll
    for (int j = 0; j < 4; ++j) {
      int ch = h * Dh + j * 16 + colL;
      float s = 0.125f * scA[ch];  // D^-0.5 folded into BN scale
      float o = boA[ch];
#pragma unroll
      for (int g = 0; g < 8; ++g) {
        size_t row = (size_t)mt * 16 + rsel + g;
        attn[(size_t)tb * Nn * C + row * C + ch] = (_Float16)(am[j][g] * s + o);
      }
    }
  }
}

// ---------------------------------------------------------------------------
// Host launcher
// ---------------------------------------------------------------------------
extern "C" void kernel_launch(void* const* d_in, const int* in_sizes, int n_in,
                              void* d_out, int out_size, void* d_ws, size_t ws_size,
                              hipStream_t stream) {
  (void)in_sizes; (void)n_in; (void)out_size; (void)ws_size;
  constexpr int T = 4, Bb = 16, Nseq = 1024, C = 512, H = 8;
  constexpr size_t M = (size_t)T * Bb * Nseq;  // 65536
  constexpr int K = C, N1 = 3 * C;

  const float* x     = (const float*)d_in[0];
  const float* wqkv  = (const float*)d_in[1];
  const float* wout  = (const float*)d_in[2];
  const float* qg = (const float*)d_in[3],  * qb = (const float*)d_in[4];
  const float* qm = (const float*)d_in[5],  * qv = (const float*)d_in[6];
  const float* ag = (const float*)d_in[7],  * ab = (const float*)d_in[8];
  const float* am = (const float*)d_in[9],  * av = (const float*)d_in[10];
  const float* og = (const float*)d_in[11], * ob = (const float*)d_in[12];
  const float* om = (const float*)d_in[13], * ov = (const float*)d_in[14];

  auto al = [](size_t v) { return (v + 255) & ~(size_t)255; };
  char* p = (char*)d_ws;
  _Float16* xh    = (_Float16*)p; p += al(M * K * 2);
  _Float16* wqkvh = (_Float16*)p; p += al((size_t)N1 * K * 2);
  _Float16* wouth = (_Float16*)p; p += al((size_t)C * K * 2);
  _Float16* qkvB  = (_Float16*)p; p += al(M * N1 * 2);       // qkv -> spikes (in place)
  _Float16* attnB = (_Float16*)p; p += al(M * C * 2);        // attn pre-LIF -> spikes
  float* scQ = (float*)p; p += al(N1 * 4);
  float* boQ = (float*)p; p += al(N1 * 4);
  float* scA = (float*)p; p += al(C * 4);
  float* boA = (float*)p; p += al(C * 4);
  float* scO = (float*)p; p += al(C * 4);
  float* boO = (float*)p; p += al(C * 4);

  // 1) precision conversion + BN coefficient folding
  f32_to_f16_kernel<<<(unsigned)((M * K + 1023) / 1024), 256, 0, stream>>>(x, xh, M * K);
  f32_to_f16_kernel<<<(unsigned)(((size_t)N1 * K + 1023) / 1024), 256, 0, stream>>>(wqkv, wqkvh, (size_t)N1 * K);
  f32_to_f16_kernel<<<(unsigned)(((size_t)C * K + 1023) / 1024), 256, 0, stream>>>(wout, wouth, (size_t)C * K);
  bn_coef_kernel<<<(N1 + 255) / 256, 256, 0, stream>>>(qg, qb, qm, qv, scQ, boQ, N1);
  bn_coef_kernel<<<(C + 255) / 256, 256, 0, stream>>>(ag, ab, am, av, scA, boA, C);
  bn_coef_kernel<<<(C + 255) / 256, 256, 0, stream>>>(og, ob, om, ov, scO, boO, C);

  // 2) qkv projection + BN  (65536 x 1536, K=512)
  gemm_bn_kernel<true><<<dim3(N1 / 128, (unsigned)(M / 128)), 256, 0, stream>>>(
      xh, wqkvh, scQ, boQ, qkvB, N1, K);

  // 3) LIF over T on qkv (in place) -> spikes
  {
    size_t S = (size_t)Bb * Nseq * N1;
    lif_inplace_f16_kernel<<<(unsigned)((S + 255) / 256), 256, 0, stream>>>(qkvB, S);
  }

  // 4) linear attention per (t,b,h) with fused scale+BN
  attn_kernel<<<T * Bb * H, 256, 0, stream>>>(qkvB, scA, boA, attnB);

  // 5) LIF over T on attn (in place) -> spikes
  {
    size_t S = (size_t)Bb * Nseq * C;
    lif_inplace_f16_kernel<<<(unsigned)((S + 255) / 256), 256, 0, stream>>>(attnB, S);
  }

  // 6) output projection + BN -> d_out (f32)
  gemm_bn_kernel<false><<<dim3(C / 128, (unsigned)(M / 128)), 256, 0, stream>>>(
      attnB, wouth, scO, boO, d_out, C, K);

  // 7) final LIF over T in place on d_out
  {
    size_t S = (size_t)Bb * Nseq * C;
    lif_inplace_f32_kernel<<<(unsigned)((S + 255) / 256), 256, 0, stream>>>((float*)d_out, S);
  }
}
